// ECFP_9457517985899
// MI455X (gfx1250) — compile-verified, matrix operand order checked by reading, not run on hardware
//
#include <hip/hip_runtime.h>
#include <hip/hip_bf16.h>

#define N_NODES 20000
#define N_EDGES 160000
#define DIM 64
#define SUB 16
#define NB 5
#define RB 8
#define NLAYERS 3
#define HID 64
#define XI_SHAPE 640          // RB*SUB*NB
#define FC_IN 656             // XI_SHAPE + SUB
#define FC_IN_PAD 672         // padded to 21 chunks of K=32
#define CUTOFF 5.0f
#define LOG2F_C 0.6931471805599453f
#define PI_F 3.14159265358979323846f

typedef __attribute__((ext_vector_type(16))) __bf16 v16bf;
typedef __attribute__((ext_vector_type(8)))  float  v8f;
typedef __attribute__((ext_vector_type(4)))  unsigned int u32x4;
typedef __attribute__((ext_vector_type(8)))  int i32x8;
typedef __attribute__((ext_vector_type(4)))  int i32x4;

// address-space qualified int4 for the async-to-LDS builtin
typedef __attribute__((address_space(1))) i32x4 g_i32x4;   // global
typedef __attribute__((address_space(3))) i32x4 l_i32x4;   // LDS

struct u4x2 { uint4 a, b; };

static __device__ inline unsigned short f2bf(float f) {
    unsigned u = __float_as_uint(f);
    unsigned r = (u + 0x7FFFu + ((u >> 16) & 1u)) >> 16;
    return (unsigned short)r;
}

static __device__ inline float dssp(float x) {
    // softplus(x) - log(2), numerically stable
    float ax = fabsf(x);
    return fmaxf(x, 0.0f) + __logf(1.0f + __expf(-ax)) - LOG2F_C;
}

static __device__ inline v16bf load_frag(const unsigned short* p) {
    // 16 bf16 = two contiguous 16B runs at +0 and +16 elements (K and K+16 halves)
    u4x2 t;
    t.a = *(const uint4*)(p);
    t.b = *(const uint4*)(p + 16);
    return __builtin_bit_cast(v16bf, t);
}

static __device__ inline v8f wmma_bf16(v16bf a, v16bf b, v8f c) {
    return __builtin_amdgcn_wmma_f32_16x16x32_bf16(false, a, false, b, (short)0, c, false, false);
}

// LDS byte offset of a __shared__ object (generic -> AS3 -> int)
#define LDS_OFF(p) ((unsigned)(unsigned long long)(__attribute__((address_space(3))) const void*)(p))

// Issue a TDM load of one contiguous row of `nelem` 2-byte elements into LDS.
// D# per CDNA5 ISA ch.8: group0 {count=1, lds_addr, global_addr, type=2},
// group1 {data_size=2B, tensor_dim0=nelem, tensor_dim1=1, tile_dim0=nelem,
//         tile_dim1=1, tensor_dim0_stride=nelem}. Groups 2/3 zero (<=2D).
static __device__ inline void tdm_load_row2b(const unsigned short* gsrc,
                                             unsigned lds_off, unsigned nelem) {
#if __has_builtin(__builtin_amdgcn_tensor_load_to_lds)
    unsigned long long ga = (unsigned long long)(size_t)gsrc;
    u32x4 g0 = { 1u,                                   // count=1, user mode
                 lds_off,                              // lds_addr
                 (unsigned)ga,                         // global_addr[31:0]
                 (unsigned)((ga >> 32) & 0x01FFFFFFu) | (2u << 30) }; // [56:32] | type=2
    i32x8 g1 = { (int)(1u << 16),                      // data_size = 1 (2 bytes)
                 (int)(nelem << 16),                   // tensor_dim0[15:0] @ bits 63:48
                 (int)((nelem >> 16) | (1u << 16)),    // tensor_dim0[31:16] | tensor_dim1=1
                 (int)(nelem << 16),                   // tile_dim0 @ bits 127:112
                 1,                                    // tile_dim1 = 1
                 (int)nelem,                           // tensor_dim0_stride[31:0]
                 0, 0 };
    i32x4 z4 = { 0, 0, 0, 0 };
#if __clang_major__ >= 23
    i32x8 z8 = { 0, 0, 0, 0, 0, 0, 0, 0 };
    __builtin_amdgcn_tensor_load_to_lds(g0, g1, z4, z4, z8, 0);
#else
    __builtin_amdgcn_tensor_load_to_lds(g0, g1, z4, z4, 0);
#endif
#else
    (void)gsrc; (void)lds_off; (void)nelem;
#endif
}

// ---------------- one-time kernels ----------------

__global__ void k_rowstart(const int* __restrict__ src, int* __restrict__ rowstart) {
    int n = blockIdx.x * 256 + threadIdx.x;
    if (n > N_NODES) return;
    int lo = 0, hi = N_EDGES;            // lower_bound(src, n)
    while (lo < hi) { int mid = (lo + hi) >> 1; if (src[mid] < n) lo = mid + 1; else hi = mid; }
    rowstart[n] = lo;
}

__global__ void k_init_xi(const int* __restrict__ species, const float* __restrict__ Wsp,
                          float* __restrict__ xi) {
    int i = blockIdx.x * 256 + threadIdx.x;
    if (i >= N_NODES * DIM) return;
    int n = i >> 6, d = i & 63;
    xi[i] = Wsp[species[n] * DIM + d];
}

// fcW1 [L][656][64] f32 -> W1T [L][64][672] bf16 (zero-padded K)
__global__ void k_w1t(const float* __restrict__ fcW1, unsigned short* __restrict__ W1T) {
    int i = blockIdx.x * 256 + threadIdx.x;
    if (i >= NLAYERS * 64 * FC_IN_PAD) return;
    int l = i / (64 * FC_IN_PAD);
    int rem = i % (64 * FC_IN_PAD);
    int n = rem / FC_IN_PAD, k = rem % FC_IN_PAD;
    float v = (k < FC_IN) ? fcW1[((size_t)l * FC_IN + k) * 64 + n] : 0.0f;
    W1T[i] = f2bf(v);
}

// W [L][64][64] f32 -> WT [L][64][64] bf16 transposed
__global__ void k_w64t(const float* __restrict__ W, unsigned short* __restrict__ WT) {
    int i = blockIdx.x * 256 + threadIdx.x;
    if (i >= NLAYERS * 64 * 64) return;
    int l = i / (64 * 64);
    int rem = i % (64 * 64);
    int n = rem / 64, k = rem % 64;
    WT[i] = f2bf(W[(l * 64 + k) * 64 + n]);
}

// ---------------- per-layer kernels ----------------

// h = xi @ Wl + bl ; si -> X[:,640:656] bf16 (and zero pad 656:672) ; mi -> f32 buffer
__global__ __launch_bounds__(128) void k_node(const float* __restrict__ xi,
                                              const float* __restrict__ Wl,
                                              const float* __restrict__ bl,
                                              float* __restrict__ mi,
                                              unsigned short* __restrict__ X) {
    __shared__ float sW[DIM * 2 * SUB];   // 64x32 = 8KB
#if __has_builtin(__builtin_amdgcn_global_load_async_to_lds_b128)
    {
        // 2048 floats = 512 x b128; 128 threads x 4 async issues
#pragma unroll
        for (int q = 0; q < 4; ++q) {
            int idx = (q * 128 + (int)threadIdx.x) * 4;
            __builtin_amdgcn_global_load_async_to_lds_b128(
                (g_i32x4*)(Wl + idx),
                (l_i32x4*)(sW + idx), 0, 0);
        }
#if __has_builtin(__builtin_amdgcn_s_wait_asynccnt)
        __builtin_amdgcn_s_wait_asynccnt(0);
#else
        asm volatile("s_wait_asynccnt 0" ::: "memory");
#endif
    }
#else
    for (int i = threadIdx.x; i < DIM * 2 * SUB; i += 128) sW[i] = Wl[i];
#endif
    __syncthreads();
    int n = blockIdx.x * 128 + threadIdx.x;
    if (n >= N_NODES) return;
    float x[DIM];
    const float4* xp = (const float4*)(xi + (size_t)n * DIM);
#pragma unroll
    for (int q = 0; q < DIM / 4; ++q) {
        float4 v = xp[q];
        x[q * 4 + 0] = v.x; x[q * 4 + 1] = v.y; x[q * 4 + 2] = v.z; x[q * 4 + 3] = v.w;
    }
    unsigned short* xrow = X + (size_t)n * FC_IN_PAD;
#pragma unroll 4
    for (int j = 0; j < 2 * SUB; ++j) {
        float s = bl[j];
#pragma unroll
        for (int k = 0; k < DIM; ++k) s = fmaf(x[k], sW[k * 2 * SUB + j], s);
        if (j < SUB) xrow[XI_SHAPE + j] = f2bf(s);   // si
        else         mi[(size_t)n * SUB + (j - SUB)] = s;
    }
#pragma unroll
    for (int j = 0; j < 16; ++j) xrow[FC_IN + j] = 0; // zero pad
}

// one wave per node; lane owns r = lane/4, s in [4*(lane&3), +4), all 5 b
__global__ __launch_bounds__(256) void k_edge(const int* __restrict__ rowstart,
                                              const int* __restrict__ edge_dst,
                                              const float* __restrict__ dist,
                                              const float* __restrict__ sw,
                                              const float* __restrict__ bo,
                                              const float* __restrict__ mi,
                                              unsigned short* __restrict__ X) {
    int gwave = (blockIdx.x * 256 + threadIdx.x) >> 5;
    int lane  = threadIdx.x & 31;
    if (gwave >= N_NODES) return;
    int n  = gwave;
    int r  = lane >> 2;           // 0..7
    int s0 = (lane & 3) * 4;      // 0,4,8,12
    float acc[4][5] = {};
    int e0 = rowstart[n], e1 = rowstart[n + 1];
    float freq = (float)(r + 1) * PI_F / CUTOFF;
    float norm = __fsqrt_rn(2.0f / CUTOFF);
    for (int e = e0; e < e1; ++e) {
        int dst = edge_dst[e];
        float x = dist[e];
        float c = sw[e];
        float rb = norm * __sinf(freq * x) / x;
        float t  = c * rb;
        float4 m4 = *(const float4*)(mi + (size_t)dst * SUB + s0);
        float bv[5];
#pragma unroll
        for (int b = 0; b < NB; ++b) bv[b] = bo[(size_t)e * NB + b];
        float ms[4] = { m4.x * t, m4.y * t, m4.z * t, m4.w * t };
#pragma unroll
        for (int s = 0; s < 4; ++s)
#pragma unroll
            for (int b = 0; b < NB; ++b) acc[s][b] = fmaf(ms[s], bv[b], acc[s][b]);
    }
    // 20 contiguous bf16 at X[n, r*80 + s0*5 ...], 8B aligned -> 5x b64 stores
    alignas(8) unsigned short out[20];
#pragma unroll
    for (int s = 0; s < 4; ++s)
#pragma unroll
        for (int b = 0; b < NB; ++b) out[s * NB + b] = f2bf(acc[s][b]);
    uint2* gp = (uint2*)(X + (size_t)n * FC_IN_PAD + r * 80 + s0 * 5);
    const uint2* op = (const uint2*)out;
#pragma unroll
    for (int i = 0; i < 5; ++i) gp[i] = op[i];
}

// fused FC1(ssp) -> FC2(ssp) -> FC3 + residual, WMMA bf16, 1 wave = 16 nodes.
// FC2/FC3 weights are streamed to LDS by the Tensor Data Mover while FC1's
// 84 WMMAs execute; the wave syncs on TENSORcnt before FC2.
__global__ __launch_bounds__(128) void k_fc(const unsigned short* __restrict__ X,
                                            const unsigned short* __restrict__ W1T,
                                            const float* __restrict__ b1,
                                            const unsigned short* __restrict__ W2T,
                                            const float* __restrict__ b2,
                                            const unsigned short* __restrict__ W3T,
                                            const float* __restrict__ b3,
                                            const float* __restrict__ xi_in,
                                            float* __restrict__ xi_out) {
    __shared__ unsigned short W2s[HID * HID];     // 8KB
    __shared__ unsigned short W3s[HID * HID];     // 8KB
    __shared__ unsigned short Hbuf[4][16 * HID];  // per-wave private H tile (bf16)
    int wave = threadIdx.x >> 5;
    int lane = threadIdx.x & 31;
    int tile = blockIdx.x * 4 + wave;
    bool active = (tile * 16 < N_NODES);          // wave-uniform at runtime
    int m0     = tile * 16;
    int mrow   = lane & 15;                       // A-fragment M / B-fragment N
    int khalf  = (lane >> 4) * 8;                 // K-half select per ISA layout
    int rowoff = (lane >> 4) * 8;                 // C/D-layout M offset

#if __has_builtin(__builtin_amdgcn_tensor_load_to_lds)
    if (threadIdx.x < 32) {                       // wave 0 drives the TDM
        tdm_load_row2b(W2T, LDS_OFF(W2s), HID * HID);
        tdm_load_row2b(W3T, LDS_OFF(W3s), HID * HID);
    }
#else
    for (int i = threadIdx.x; i < HID * HID; i += 128) { W2s[i] = W2T[i]; W3s[i] = W3T[i]; }
#endif

    // ---- FC1: [16 x 672] @ [672 x 64], B from L2-resident global ----
    v8f acc[4] = {};
    const unsigned short* arow = X + (size_t)(m0 + mrow) * FC_IN_PAD + khalf;
    if (active) {
        for (int kc = 0; kc < FC_IN_PAD; kc += 32) {
            __builtin_prefetch(arow + kc + 64, 0, 3);
            v16bf a = load_frag(arow + kc);
#pragma unroll
            for (int nt = 0; nt < 4; ++nt) {
                v16bf b = load_frag(W1T + (size_t)(nt * 16 + mrow) * FC_IN_PAD + kc + khalf);
                acc[nt] = wmma_bf16(a, b, acc[nt]);
            }
        }
        unsigned short* hb = &Hbuf[wave][0];
#pragma unroll
        for (int nt = 0; nt < 4; ++nt) {
            float bias = b1[nt * 16 + mrow];
#pragma unroll
            for (int r = 0; r < 8; ++r) {
                float v = dssp(acc[nt][r] + bias);
                hb[(r + rowoff) * HID + nt * 16 + mrow] = f2bf(v);
            }
        }
    }

#if __has_builtin(__builtin_amdgcn_tensor_load_to_lds)
    if (threadIdx.x < 32) __builtin_amdgcn_s_wait_tensorcnt(0);
#endif
    __syncthreads();                              // W2s/W3s visible to all waves

    if (active) {
        unsigned short* hb = &Hbuf[wave][0];
        const unsigned short* hrow = hb + mrow * HID + khalf;

        // ---- FC2: [16 x 64] @ [64 x 64], B from LDS ----
        v8f acc2[4] = {};
#pragma unroll
        for (int kc = 0; kc < HID; kc += 32) {
            v16bf a = load_frag(hrow + kc);
#pragma unroll
            for (int nt = 0; nt < 4; ++nt) {
                v16bf b = load_frag(W2s + (size_t)(nt * 16 + mrow) * HID + kc + khalf);
                acc2[nt] = wmma_bf16(a, b, acc2[nt]);
            }
        }
#pragma unroll
        for (int nt = 0; nt < 4; ++nt) {
            float bias = b2[nt * 16 + mrow];
#pragma unroll
            for (int r = 0; r < 8; ++r) {
                float v = dssp(acc2[nt][r] + bias);
                hb[(r + rowoff) * HID + nt * 16 + mrow] = f2bf(v);
            }
        }

        // ---- FC3: [16 x 64] @ [64 x 64] + bias + residual, B from LDS ----
        v8f acc3[4] = {};
#pragma unroll
        for (int kc = 0; kc < HID; kc += 32) {
            v16bf a = load_frag(hrow + kc);
#pragma unroll
            for (int nt = 0; nt < 4; ++nt) {
                v16bf b = load_frag(W3s + (size_t)(nt * 16 + mrow) * HID + kc + khalf);
                acc3[nt] = wmma_bf16(a, b, acc3[nt]);
            }
        }
#pragma unroll
        for (int nt = 0; nt < 4; ++nt) {
            float bias = b3[nt * 16 + mrow];
            int col = nt * 16 + mrow;
#pragma unroll
            for (int r = 0; r < 8; ++r) {
                int node = m0 + r + rowoff;
                float v = acc3[nt][r] + bias + xi_in[(size_t)node * DIM + col];
                xi_out[(size_t)node * DIM + col] = v;
            }
        }
    }
}

// ---------------- host launcher ----------------

extern "C" void kernel_launch(void* const* d_in, const int* in_sizes, int n_in,
                              void* d_out, int out_size, void* d_ws, size_t ws_size,
                              hipStream_t stream) {
    (void)in_sizes; (void)n_in; (void)out_size; (void)ws_size;
    const int*   species   = (const int*)d_in[0];
    const int*   edge_src  = (const int*)d_in[1];
    const int*   edge_dst  = (const int*)d_in[2];
    const float* distances = (const float*)d_in[3];
    const float* swv       = (const float*)d_in[4];
    const float* bo        = (const float*)d_in[5];
    const float* W_species = (const float*)d_in[6];
    const float* Wl        = (const float*)d_in[7];
    const float* bl        = (const float*)d_in[8];
    const float* fcW1      = (const float*)d_in[9];
    const float* fcb1      = (const float*)d_in[10];
    const float* fcW2      = (const float*)d_in[11];
    const float* fcb2      = (const float*)d_in[12];
    const float* fcW3      = (const float*)d_in[13];
    const float* fcb3      = (const float*)d_in[14];
    float* out = (float*)d_out;

    char* base = (char*)d_ws;
    size_t off = 0;
    auto carve = [&](size_t bytes) -> char* {
        char* p = base + off;
        off = (off + bytes + 255) & ~(size_t)255;
        return p;
    };
    float*          xiA  = (float*)carve((size_t)N_NODES * DIM * 4);
    float*          xiB  = (float*)carve((size_t)N_NODES * DIM * 4);
    float*          mi   = (float*)carve((size_t)N_NODES * SUB * 4);
    unsigned short* X    = (unsigned short*)carve((size_t)N_NODES * FC_IN_PAD * 2);
    int*            rows = (int*)carve((size_t)(N_NODES + 1) * 4);
    unsigned short* W1T  = (unsigned short*)carve((size_t)NLAYERS * 64 * FC_IN_PAD * 2);
    unsigned short* W2T  = (unsigned short*)carve((size_t)NLAYERS * 64 * 64 * 2);
    unsigned short* W3T  = (unsigned short*)carve((size_t)NLAYERS * 64 * 64 * 2);

    // one-time prep
    k_rowstart<<<(N_NODES + 1 + 255) / 256, 256, 0, stream>>>(edge_src, rows);
    k_w1t<<<(NLAYERS * 64 * FC_IN_PAD + 255) / 256, 256, 0, stream>>>(fcW1, W1T);
    k_w64t<<<(NLAYERS * 64 * 64 + 255) / 256, 256, 0, stream>>>(fcW2, W2T);
    k_w64t<<<(NLAYERS * 64 * 64 + 255) / 256, 256, 0, stream>>>(fcW3, W3T);
    k_init_xi<<<(N_NODES * DIM + 255) / 256, 256, 0, stream>>>(species, W_species, xiA);

    const int fc_blocks   = (N_NODES / 16 + 3) / 4;          // 4 waves/block, 16 nodes/wave
    const int edge_blocks = (N_NODES * 32 + 255) / 256;      // 1 wave/node
    const int node_blocks = (N_NODES + 127) / 128;

    float* xi_in = xiA;
    for (int l = 0; l < NLAYERS; ++l) {
        float* xi_out = (l == NLAYERS - 1) ? out : ((xi_in == xiA) ? xiB : xiA);
        k_node<<<node_blocks, 128, 0, stream>>>(xi_in,
                                                Wl + (size_t)l * DIM * 2 * SUB,
                                                bl + (size_t)l * 2 * SUB, mi, X);
        k_edge<<<edge_blocks, 256, 0, stream>>>(rows, edge_dst, distances, swv, bo, mi, X);
        k_fc<<<fc_blocks, 128, 0, stream>>>(X,
                                            W1T + (size_t)l * 64 * FC_IN_PAD, fcb1 + (size_t)l * 64,
                                            W2T + (size_t)l * 64 * 64,        fcb2 + (size_t)l * 64,
                                            W3T + (size_t)l * 64 * 64,        fcb3 + (size_t)l * 64,
                                            xi_in, xi_out);
        xi_in = xi_out;
    }
}
